// ObjectDecoder_45861660787131
// MI455X (gfx1250) — matrix operand order repaired and checked by reading, not compiled, not run on hardware
//
#include <hip/hip_runtime.h>
#include <cstdint>
#include <cstddef>

// ---------------------------------------------------------------------------
// Feature detection: CDNA5 async global->LDS copy
// ---------------------------------------------------------------------------
#if defined(__has_builtin)
#  if __has_builtin(__builtin_amdgcn_global_load_async_to_lds_b128)
#    define USE_ASYNC_LDS 1
#  endif
#endif
#ifndef USE_ASYNC_LDS
#  define USE_ASYNC_LDS 0
#endif

// Builtin parameter types (from hipcc diagnostic):
//   arg0: AS1 pointer to int __attribute__((vector_size(16)))
//   arg1: AS3 pointer to same; arg2/arg3: imm offset / cpol
typedef __attribute__((__vector_size__(4 * sizeof(int)))) int v4i_b128;
typedef __attribute__((address_space(1))) v4i_b128* glob_b128_ptr;
typedef __attribute__((address_space(3))) v4i_b128* lds_b128_ptr;

__device__ __forceinline__ void wait_async_all() {
#if USE_ASYNC_LDS
#  if __has_builtin(__builtin_amdgcn_s_wait_asynccnt)
  __builtin_amdgcn_s_wait_asynccnt(0);
#  else
  asm volatile("s_wait_asynccnt 0x0" ::: "memory");
#  endif
#endif
}

// ---------------------------------------------------------------------------
// Types / helpers
// ---------------------------------------------------------------------------
typedef __attribute__((ext_vector_type(16))) __bf16 v16bf;
typedef __attribute__((ext_vector_type(8)))  float  v8f;

struct alignas(16) U128 { uint32_t a, b, c, d; };

__device__ __forceinline__ uint16_t f32_to_bf16(float f) {
  union { float f; uint32_t u; } x; x.f = f;
  uint32_t u = x.u;
  return (uint16_t)((u + 0x7FFFu + ((u >> 16) & 1u)) >> 16);
}

__device__ __forceinline__ float block_sum(float v, float* red) {
  int t = threadIdx.x;
  red[t] = v; __syncthreads();
  #pragma unroll
  for (int s = 128; s > 0; s >>= 1) {
    if (t < s) red[t] += red[t + s];
    __syncthreads();
  }
  float r = red[0];
  __syncthreads();
  return r;
}

__device__ __forceinline__ float block_max(float v, float* red) {
  int t = threadIdx.x;
  red[t] = v; __syncthreads();
  #pragma unroll
  for (int s = 128; s > 0; s >>= 1) {
    if (t < s) red[t] = fmaxf(red[t], red[t + s]);
    __syncthreads();
  }
  float r = red[0];
  __syncthreads();
  return r;
}

// ---------------------------------------------------------------------------
// Tiled bf16 WMMA GEMM:  C[M,N] = A[M,K] * Bt[N,K]^T (+bias/+resid), fp32 accum.
// B is supplied PRE-TRANSPOSED (Bt row-major [N][K]) so both A and B staging
// are plain row-major copies -> async global->LDS b128 copies, no registers.
// Block tile 128x128, BK=64; 4 waves (128 thr), each wave a 64x64 tile =
// 4x4 WMMA accumulators -> 32 v_wmma_f32_16x16x32_bf16 per K-stage per wave.
// Double-buffered LDS; one s_wait_asynccnt + one barrier per stage.
// Requires M % 128 == 0, N % 128 == 0, K % 64 == 0 (true for all calls here).
// ---------------------------------------------------------------------------
enum { FB_BIAS = 1, FB_WF32 = 2, FB_WBF16 = 4, FB_TRANS = 8, FB_RESID = 16 };

template<int FLAGS>
__global__ __launch_bounds__(128, 1)
void gemm_wmma_bf16(const uint16_t* __restrict__ A,
                    const uint16_t* __restrict__ Bt,
                    const float* __restrict__ bias,
                    const float* __restrict__ resid,
                    float* __restrict__ outF,
                    uint16_t* __restrict__ outH,
                    int Mrows, int Ncols, int K)
{
  // stride 72 halves (= 144 B, mult of 16): aligned b128 frags, no bank clash
  __shared__ alignas(16) uint16_t ldsA[2][128 * 72];   // [row][k]
  __shared__ alignas(16) uint16_t ldsB[2][128 * 72];   // [col][k]

  const int tid  = threadIdx.x;
  const int lane = tid & 31;
  const int wave = tid >> 5;
  const int wm   = wave >> 1;   // 0..1  (64-row strip)
  const int wn   = wave & 1;    // 0..1  (64-col strip)
  const int ln15 = lane & 15;
  const int ksel = lane >> 4;   // 0/1

  const int blockRow = blockIdx.y * 128;
  const int blockCol = blockIdx.x * 128;

  v8f acc[4][4] = {};

  // Stage one 128x64-half panel of A and of Bt into LDS buffer `buf`.
  // 8 b128 copies per thread per matrix (128 threads x 16B = 2KB/unit).
  auto issueFills = [&](int kb, int buf) {
    #pragma unroll
    for (int u = 0; u < 8; ++u) {
      int idx = tid + u * 128;
      int r = idx >> 3, c = idx & 7;          // row 0..127, b128 col 0..7
      const uint16_t* ga = A  + (size_t)(blockRow + r) * K + kb + c * 8;
      const uint16_t* gb = Bt + (size_t)(blockCol + r) * K + kb + c * 8;
      uint16_t* la = &ldsA[buf][r * 72 + c * 8];
      uint16_t* lb = &ldsB[buf][r * 72 + c * 8];
#if USE_ASYNC_LDS
      __builtin_amdgcn_global_load_async_to_lds_b128(
          (glob_b128_ptr)ga, (lds_b128_ptr)la, 0, 0);
      __builtin_amdgcn_global_load_async_to_lds_b128(
          (glob_b128_ptr)gb, (lds_b128_ptr)lb, 0, 0);
#else
      *(U128*)la = *(const U128*)ga;
      *(U128*)lb = *(const U128*)gb;
#endif
    }
  };

  issueFills(0, 0);
  const int nStages = K >> 6;
  for (int s = 0; s < nStages; ++s) {
    const int cur = s & 1;
    wait_async_all();       // this wave's fills of `cur` have landed in LDS
    __syncthreads();        // all waves' fills visible; prev reads finished
    if (s + 1 < nStages) issueFills((s + 1) << 6, cur ^ 1);  // fly over compute

    union Frag { U128 u[2]; v16bf v; };
    #pragma unroll
    for (int kk = 0; kk < 2; ++kk) {
      Frag af[4], bf[4];
      #pragma unroll
      for (int ti = 0; ti < 4; ++ti) {
        // A layout: lane = row m; halves 0..7 = K 8k..8k+7, 8..15 = K 16+8k..
        int row = wm * 64 + ti * 16 + ln15;
        const uint16_t* base = &ldsA[cur][row * 72 + kk * 32 + 8 * ksel];
        af[ti].u[0] = *(const U128*)base;
        af[ti].u[1] = *(const U128*)(base + 16);
      }
      #pragma unroll
      for (int tj = 0; tj < 4; ++tj) {
        // B layout: lane = col n; halves 0..7 = K 8k..8k+7, 8..15 = K 16+8k..
        int col = wn * 64 + tj * 16 + ln15;
        const uint16_t* base = &ldsB[cur][col * 72 + kk * 32 + 8 * ksel];
        bf[tj].u[0] = *(const U128*)base;
        bf[tj].u[1] = *(const U128*)(base + 16);
      }
      #pragma unroll
      for (int ti = 0; ti < 4; ++ti)
        #pragma unroll
        for (int tj = 0; tj < 4; ++tj)
          acc[ti][tj] = __builtin_amdgcn_wmma_f32_16x16x32_bf16(
              false, af[ti].v, false, bf[tj].v, (short)0, acc[ti][tj], false, false);
    }
  }

  // ---- epilogue: C/D layout -> n = lane&15, m = r + 8*(lane>>4) ----
  #pragma unroll
  for (int ti = 0; ti < 4; ++ti) {
    #pragma unroll
    for (int tj = 0; tj < 4; ++tj) {
      int col = blockCol + wn * 64 + tj * 16 + ln15;
      float bia = (FLAGS & FB_BIAS) ? bias[col] : 0.0f;
      #pragma unroll
      for (int r = 0; r < 8; ++r) {
        int row = blockRow + wm * 64 + ti * 16 + r + 8 * ksel;
        float v = acc[ti][tj][r] + bia;
        if (FLAGS & FB_RESID) v += resid[(size_t)row * Ncols + col];
        if (FLAGS & FB_WF32)  outF[(size_t)row * Ncols + col] = v;
        if (FLAGS & FB_WBF16) {
          uint16_t h = f32_to_bf16(v);
          if (FLAGS & FB_TRANS) outH[(size_t)col * Mrows + row] = h;
          else                  outH[(size_t)row * Ncols + col] = h;
        }
      }
    }
  }
}

// ---------------------------------------------------------------------------
// fp32 [rows][cols] -> bf16 transposed [cols][rows] (tiled)
// ---------------------------------------------------------------------------
__global__ __launch_bounds__(256)
void cast_transpose_f32_bf16(const float* __restrict__ in,
                             uint16_t* __restrict__ out, int rows, int cols)
{
  __shared__ uint16_t tile[32][33];
  int c0 = blockIdx.x * 32, r0 = blockIdx.y * 32;
  int tx = threadIdx.x & 31, ty = threadIdx.x >> 5;   // 32 x 8
  #pragma unroll
  for (int i = 0; i < 4; ++i) {
    int r = ty + i * 8;
    tile[r][tx] = f32_to_bf16(in[(size_t)(r0 + r) * cols + c0 + tx]);
  }
  __syncthreads();
  #pragma unroll
  for (int i = 0; i < 4; ++i) {
    int r = ty + i * 8;
    out[(size_t)(c0 + r) * rows + r0 + tx] = tile[tx][r];
  }
}

// ---------------------------------------------------------------------------
// Row LayerNorm over D=1024 (one row per block, 256 threads x 4 elems)
// ---------------------------------------------------------------------------
template<bool WF32, bool WBF16>
__global__ __launch_bounds__(256)
void ln_rows(const float* __restrict__ x, const float* __restrict__ g,
             const float* __restrict__ bt,
             float* __restrict__ outF, uint16_t* __restrict__ outH)
{
  __shared__ float red[256];
  size_t row = blockIdx.x;
  const float* xr = x + row * 1024;
  float v[4]; float s = 0.0f;
  #pragma unroll
  for (int i = 0; i < 4; ++i) { v[i] = xr[threadIdx.x + i * 256]; s += v[i]; }
  float mu = block_sum(s, red) * (1.0f / 1024.0f);
  float sq = 0.0f;
  #pragma unroll
  for (int i = 0; i < 4; ++i) { float d = v[i] - mu; sq += d * d; }
  float var = block_sum(sq, red) * (1.0f / 1024.0f);
  float rstd = rsqrtf(var + 1e-5f);
  #pragma unroll
  for (int i = 0; i < 4; ++i) {
    int c = threadIdx.x + i * 256;
    float y = (v[i] - mu) * rstd * g[c] + bt[c];
    if (WF32)  outF[row * 1024 + c] = y;
    if (WBF16) outH[row * 1024 + c] = f32_to_bf16(y);
  }
}

// LayerNorm + exact GELU; optionally add residual and emit fp32, else bf16.
template<bool ADD_RESID>
__global__ __launch_bounds__(256)
void ln_gelu_rows(const float* __restrict__ x, const float* __restrict__ g,
                  const float* __restrict__ bt, const float* __restrict__ resid,
                  float* __restrict__ outF, uint16_t* __restrict__ outH)
{
  __shared__ float red[256];
  size_t row = blockIdx.x;
  const float* xr = x + row * 1024;
  float v[4]; float s = 0.0f;
  #pragma unroll
  for (int i = 0; i < 4; ++i) { v[i] = xr[threadIdx.x + i * 256]; s += v[i]; }
  float mu = block_sum(s, red) * (1.0f / 1024.0f);
  float sq = 0.0f;
  #pragma unroll
  for (int i = 0; i < 4; ++i) { float d = v[i] - mu; sq += d * d; }
  float var = block_sum(sq, red) * (1.0f / 1024.0f);
  float rstd = rsqrtf(var + 1e-5f);
  #pragma unroll
  for (int i = 0; i < 4; ++i) {
    int c = threadIdx.x + i * 256;
    float y = (v[i] - mu) * rstd * g[c] + bt[c];
    float ge = 0.5f * y * (1.0f + erff(y * 0.70710678118654752f));
    if (ADD_RESID) outF[row * 1024 + c] = resid[row * 1024 + c] + ge;
    else           outH[row * 1024 + c] = f32_to_bf16(ge);
  }
}

// Softmax over M=256 (one row per block, 256 threads)
__global__ __launch_bounds__(256)
void softmax_rows(const float* __restrict__ logits, uint16_t* __restrict__ probs)
{
  __shared__ float red[256];
  size_t base = (size_t)blockIdx.x * 256;
  float e = logits[base + threadIdx.x];
  float mx = block_max(e, red);
  float p = __expf(e - mx);
  float s = block_sum(p, red);
  probs[base + threadIdx.x] = f32_to_bf16(p / s);
}

// ---------------------------------------------------------------------------
// BCIM: pixel vector x_p = concat(p3[b, pix, :], p3[b, 256+pix, :]) (2048 elems)
// rnorm[b,pix] = 1/||x_p||;  sim = rnorm_p/9 * sum_e x_p[e] * (sum_nbr rnorm_q * x_q[e])
// out = p3 * sim[pixel]
// ---------------------------------------------------------------------------
__global__ __launch_bounds__(256)
void bcim_rnorm(const float* __restrict__ p3, float* __restrict__ rnorm)
{
  __shared__ float red[256];
  int b = blockIdx.x >> 8, pix = blockIdx.x & 255;
  size_t r0 = ((size_t)b * 512 + pix) * 1024;
  size_t r1 = r0 + (size_t)256 * 1024;
  float s = 0.0f;
  #pragma unroll
  for (int i = 0; i < 4; ++i) {
    int d = threadIdx.x + i * 256;
    float a = p3[r0 + d], c = p3[r1 + d];
    s += a * a + c * c;
  }
  float tot = block_sum(s, red);
  if (threadIdx.x == 0) rnorm[blockIdx.x] = rsqrtf(tot);
}

__global__ __launch_bounds__(256)
void bcim_sim(const float* __restrict__ p3, const float* __restrict__ rnorm,
              float* __restrict__ sim)
{
  __shared__ float red[256];
  int b = blockIdx.x >> 8, pix = blockIdx.x & 255;
  int h = pix >> 4, w = pix & 15;
  int npix[9]; float nrq[9]; int cnt = 0;
  #pragma unroll
  for (int dh = -1; dh <= 1; ++dh)
    #pragma unroll
    for (int dw = -1; dw <= 1; ++dw) {
      int h2 = h + dh, w2 = w + dw;
      if (h2 >= 0 && h2 < 16 && w2 >= 0 && w2 < 16) {
        npix[cnt] = h2 * 16 + w2;
        nrq[cnt]  = rnorm[(b << 8) + npix[cnt]];
        ++cnt;
      }
    }
  float acc = 0.0f;
  for (int s = 0; s < 2; ++s) {
    size_t rowbase = ((size_t)b * 512 + (size_t)s * 256) * 1024;
    #pragma unroll
    for (int i = 0; i < 4; ++i) {
      int d = threadIdx.x + i * 256;
      float xp = p3[rowbase + (size_t)pix * 1024 + d];
      float t = 0.0f;
      for (int j = 0; j < cnt; ++j)
        t += nrq[j] * p3[rowbase + (size_t)npix[j] * 1024 + d];
      acc += xp * t;
    }
  }
  float tot = block_sum(acc, red);
  if (threadIdx.x == 0)
    sim[blockIdx.x] = rnorm[blockIdx.x] * tot * (1.0f / 9.0f);
}

__global__ void bcim_apply(const float* __restrict__ p3,
                           const float* __restrict__ sim,
                           float* __restrict__ out, long long n)
{
  long long i = (long long)blockIdx.x * blockDim.x + threadIdx.x;
  long long stride = (long long)gridDim.x * blockDim.x;
  for (; i < n; i += stride) {
    long long nrow = i >> 10;               // flat row index [0, 32768)
    int b   = (int)(nrow >> 9);
    int pix = (int)(nrow & 255);            // (n % 512) % 256
    out[i] = p3[i] * sim[(b << 8) + pix];
  }
}

// ---------------------------------------------------------------------------
// Host orchestration
// ---------------------------------------------------------------------------
extern "C" void kernel_launch(void* const* d_in, const int* in_sizes, int n_in,
                              void* d_out, int out_size, void* d_ws, size_t ws_size,
                              hipStream_t stream)
{
  (void)in_sizes; (void)n_in; (void)ws_size;

  const float* p_vec = (const float*)d_in[0];
  const float* o_vec = (const float*)d_in[1];
  const float* ln_p_g = (const float*)d_in[2];
  const float* ln_p_b = (const float*)d_in[3];
  const float* ln_o_g = (const float*)d_in[4];
  const float* ln_o_b = (const float*)d_in[5];
  const float* Wq = (const float*)d_in[6];  const float* bq = (const float*)d_in[7];
  const float* Wk = (const float*)d_in[8];  const float* bk = (const float*)d_in[9];
  const float* Wv = (const float*)d_in[10]; const float* bv = (const float*)d_in[11];
  const float* W1 = (const float*)d_in[12]; const float* b1 = (const float*)d_in[13];
  const float* ln1_g = (const float*)d_in[14]; const float* ln1_b = (const float*)d_in[15];
  const float* W2 = (const float*)d_in[16]; const float* b2 = (const float*)d_in[17];
  const float* ln2_g = (const float*)d_in[18]; const float* ln2_b = (const float*)d_in[19];
  float* out = (float*)d_out;

  const size_t R = 32768;   // B*N rows
  const size_t D = 1024;
  char* ws = (char*)d_ws;
  size_t off = 0;
  auto alloc = [&](size_t bytes) -> char* {
    char* p = ws + off;
    off += (bytes + 255) & ~(size_t)255;
    return p;
  };

  // weights stored TRANSPOSED ([N][K] bf16) for the BT-GEMM
  uint16_t* WqT  = (uint16_t*)alloc(D * D * 2);
  uint16_t* WkT  = (uint16_t*)alloc(D * D * 2);
  uint16_t* WvT  = (uint16_t*)alloc(D * D * 2);
  uint16_t* W1T  = (uint16_t*)alloc(D * D * 2);
  uint16_t* W2T  = (uint16_t*)alloc(D * D * 2);
  uint16_t* oln_h = (uint16_t*)alloc(256 * D * 2);
  uint16_t* k_h   = (uint16_t*)alloc(256 * D * 2);   // k row-major = Bt of logits
  uint16_t* vT_h  = (uint16_t*)alloc(D * 256 * 2);   // v^T = Bt of attn-out
  float*    pln_f = (float*)   alloc(R * D * 4);
  uint16_t* pln_h = (uint16_t*)alloc(R * D * 2);
  uint16_t* q_h   = (uint16_t*)alloc(R * D * 2);
  float*    logit = (float*)   alloc(R * 256 * 4);
  uint16_t* A_h   = (uint16_t*)alloc(R * 256 * 2);
  float*    p2_f  = (float*)   alloc(R * D * 4);
  uint16_t* p2_h  = (uint16_t*)alloc(R * D * 2);
  float*    hraw  = (float*)   alloc(R * D * 4);
  uint16_t* h_h   = (uint16_t*)alloc(R * D * 2);
  float*    rn    = (float*)   alloc(64 * 256 * 4);
  float*    simv  = (float*)   alloc(64 * 256 * 4);
  float*    p3_f  = pln_f;     // reuse: pln_f dead after attention residual read

  // weight cast+transpose
  dim3 tg(32, 32);
  cast_transpose_f32_bf16<<<tg, 256, 0, stream>>>(Wq, WqT, 1024, 1024);
  cast_transpose_f32_bf16<<<tg, 256, 0, stream>>>(Wk, WkT, 1024, 1024);
  cast_transpose_f32_bf16<<<tg, 256, 0, stream>>>(Wv, WvT, 1024, 1024);
  cast_transpose_f32_bf16<<<tg, 256, 0, stream>>>(W1, W1T, 1024, 1024);
  cast_transpose_f32_bf16<<<tg, 256, 0, stream>>>(W2, W2T, 1024, 1024);

  // LayerNorms
  ln_rows<true, true><<<32768, 256, 0, stream>>>(p_vec, ln_p_g, ln_p_b, pln_f, pln_h);
  ln_rows<false, true><<<256, 256, 0, stream>>>(o_vec, ln_o_g, ln_o_b, nullptr, oln_h);

  // k = o_ln@Wk + bk (row-major, serves as Bt for logits); v^T via TRANS epi
  gemm_wmma_bf16<FB_BIAS | FB_WBF16>
      <<<dim3(8, 2), 128, 0, stream>>>(oln_h, WkT, bk, nullptr, nullptr, k_h, 256, 1024, 1024);
  gemm_wmma_bf16<FB_BIAS | FB_WBF16 | FB_TRANS>
      <<<dim3(8, 2), 128, 0, stream>>>(oln_h, WvT, bv, nullptr, nullptr, vT_h, 256, 1024, 1024);

  // q = p_ln@Wq + bq
  gemm_wmma_bf16<FB_BIAS | FB_WBF16>
      <<<dim3(8, 256), 128, 0, stream>>>(pln_h, WqT, bq, nullptr, nullptr, q_h, 32768, 1024, 1024);

  // logits = q @ k^T   (Bt = k)
  gemm_wmma_bf16<FB_WF32>
      <<<dim3(2, 256), 128, 0, stream>>>(q_h, k_h, nullptr, nullptr, logit, nullptr, 32768, 256, 1024);

  softmax_rows<<<32768, 256, 0, stream>>>(logit, A_h);

  // p2 = p_ln + A @ v   (Bt = v^T)
  gemm_wmma_bf16<FB_RESID | FB_WF32 | FB_WBF16>
      <<<dim3(8, 256), 128, 0, stream>>>(A_h, vT_h, nullptr, pln_f, p2_f, p2_h, 32768, 1024, 256);

  // h = gelu(ln(p2@W1 + b1))
  gemm_wmma_bf16<FB_BIAS | FB_WF32>
      <<<dim3(8, 256), 128, 0, stream>>>(p2_h, W1T, b1, nullptr, hraw, nullptr, 32768, 1024, 1024);
  ln_gelu_rows<false><<<32768, 256, 0, stream>>>(hraw, ln1_g, ln1_b, nullptr, nullptr, h_h);

  // p3 = p2 + gelu(ln(h@W2 + b2))
  gemm_wmma_bf16<FB_BIAS | FB_WF32>
      <<<dim3(8, 256), 128, 0, stream>>>(h_h, W2T, b2, nullptr, hraw, nullptr, 32768, 1024, 1024);
  ln_gelu_rows<true><<<32768, 256, 0, stream>>>(hraw, ln2_g, ln2_b, p2_f, p3_f, nullptr);

  // BCIM
  bcim_rnorm<<<64 * 256, 256, 0, stream>>>(p3_f, rn);
  bcim_sim<<<64 * 256, 256, 0, stream>>>(p3_f, rn, simv);
  bcim_apply<<<4096, 256, 0, stream>>>(p3_f, simv, out, (long long)out_size);
}
